// LSTMModel_74002286510278
// MI455X (gfx1250) — compile-verified, matrix operand order checked by reading, not run on hardware
//
#include <hip/hip_runtime.h>
#include <math.h>

// Problem constants (match reference)
#define B_ 256
#define T_ 1024
#define D_ 32
#define H_ 64
#define G_ 256   // 4*H gates (i,f,g,o)

typedef __attribute__((ext_vector_type(16))) _Float16 v16h;
typedef __attribute__((ext_vector_type(8)))  _Float16 v8h;
typedef __attribute__((ext_vector_type(8)))  float    v8f;

// Branch-free hardware transcendentals: v_exp_f32 is exp2, so
// sigmoid(x) = rcp(1 + exp2(-x*log2(e))), tanh(x) = 2*sigmoid(2x) - 1.
// Keeps EXEC all-1s through the whole scan loop (no libm divergent paths).
__device__ __forceinline__ float fsig(float x) {
    float e = __builtin_amdgcn_exp2f(x * -1.442695040888963f);
    return __builtin_amdgcn_rcpf(1.0f + e);
}
__device__ __forceinline__ float ftanh(float x) {
    float e = __builtin_amdgcn_exp2f(x * -2.885390081777927f);
    return __builtin_amdgcn_rcpf(1.0f + e) * 2.0f - 1.0f;
}

// Build a 16x16x32 WMMA B-fragment (K x N, f16) for this lane by gathering
// from an f32 weight row W[n][k] (row pointer already offset to gate row n).
// Element e maps to k = korigin + (e>>3)*16 + (e&7), per the CDNA5 16-bit
// matrix VGPR striping (korigin = kf*32 + (lane>>4)*8).
__device__ __forceinline__ v16h load_bfrag(const float* __restrict__ Wrow, int korigin) {
    v16h b;
#pragma unroll
    for (int e = 0; e < 16; ++e) {
        int k = korigin + ((e >> 3) << 4) + (e & 7);
        b[e] = (_Float16)Wrow[k];
    }
    return b;
}

// A-fragment (16x32 f16, M x K) from an f16 buffer (LDS h-state).
// Elements 0..7 are k korigin..korigin+7 (one 16B ds_load_b128 half),
// elements 8..15 are at +16.
__device__ __forceinline__ v16h load_afrag_h(const _Float16* p) {
    v8h lo = *(const v8h*)(p);
    v8h hi = *(const v8h*)(p + 16);
    return __builtin_shufflevector(lo, hi, 0, 1, 2, 3, 4, 5, 6, 7,
                                           8, 9, 10, 11, 12, 13, 14, 15);
}

// A-fragment gathered from f32 global (x rows), converted inline to f16
// (compiler vectorizes to global_load_b128 + v_cvt_pk_f16_f32).
__device__ __forceinline__ v16h load_afrag_f32(const float* __restrict__ p) {
    v16h a;
#pragma unroll
    for (int e = 0; e < 16; ++e) {
        a[e] = (_Float16)p[((e >> 3) << 4) + (e & 7)];
    }
    return a;
}

__device__ __forceinline__ v8f wmma_f16(v16h a, v16h b, v8f c) {
    return __builtin_amdgcn_wmma_f32_16x16x32_f16(
        /*neg_a=*/false, a, /*neg_b=*/false, b,
        /*c_mod=*/(short)0, c, /*reuse_a=*/false, /*reuse_b=*/false);
}

// Fused 2-layer LSTM scan + FC head.
// Grid: B/16 = 16 blocks. Block: 512 threads = 16 waves (wave32).
// Each block owns 16 batch rows; wave w owns gate columns [w*16, w*16+16).
// Per step, per wave: 7 x v_wmma_f32_16x16x32_f16.
__global__ void __launch_bounds__(512, 1) lstm2_scan_kernel(
    const float* __restrict__ x,
    const float* __restrict__ w_ih0, const float* __restrict__ w_hh0,
    const float* __restrict__ b_ih0, const float* __restrict__ b_hh0,
    const float* __restrict__ w_ih1, const float* __restrict__ w_hh1,
    const float* __restrict__ b_ih1, const float* __restrict__ b_hh1,
    const float* __restrict__ w_fc,  const float* __restrict__ b_fc,
    float* __restrict__ out)
{
    __shared__ __align__(16) _Float16 h1h[16 * H_];   // layer-1 hidden (f16, A-matrix source)
    __shared__ __align__(16) _Float16 h2h[16 * H_];   // layer-2 hidden
    __shared__ __align__(16) float    gates[16 * G_]; // raw gate pre-activations

    const int tid = threadIdx.x;
    const int w   = tid >> 5;        // wave id 0..15 -> gate N-tile
    const int L   = tid & 31;        // lane
    const int b0  = blockIdx.x * 16; // batch tile base

    const int m        = L & 15;           // A row / D column within tile
    const int laneBase = (L >> 4) << 3;    // 0 or 8 (K / M-row offset)
    const int n        = (w << 4) + m;     // global gate column owned by this lane

    // Zero hidden state
    for (int i = tid; i < 16 * H_; i += blockDim.x) {
        h1h[i] = (_Float16)0.f;
        h2h[i] = (_Float16)0.f;
    }

    // Weight B-fragments, resident in VGPRs for the whole scan.
    const v16h Bx   = load_bfrag(w_ih0 + n * D_, laneBase);        // K=32 (=D)
    const v16h Bh0a = load_bfrag(w_hh0 + n * H_, laneBase);        // K 0..31
    const v16h Bh0b = load_bfrag(w_hh0 + n * H_, 32 + laneBase);   // K 32..63
    const v16h Bi1a = load_bfrag(w_ih1 + n * H_, laneBase);
    const v16h Bi1b = load_bfrag(w_ih1 + n * H_, 32 + laneBase);
    const v16h Bh1a = load_bfrag(w_hh1 + n * H_, laneBase);
    const v16h Bh1b = load_bfrag(w_hh1 + n * H_, 32 + laneBase);

    const float bias0 = b_ih0[n] + b_hh0[n];
    const float bias1 = b_ih1[n] + b_hh1[n];

    // Per-thread cell state: wave w's threads own batch row mu=w,
    // lane L owns hidden units hid = 2L, 2L+1.
    const int mu   = tid >> 5;
    const int hid0 = (tid & 31) << 1;
    float c1[2] = {0.f, 0.f};
    float c2[2] = {0.f, 0.f};

    __syncthreads();

    const float* xrow = x + ((size_t)(b0 + m) * T_) * D_ + laneBase;

    for (int t = 0; t < T_; ++t) {
        // -------- layer 0: gates = bias + x_t @ Wih0^T + h1 @ Whh0^T --------
        __builtin_prefetch(xrow + D_, 0, 1);        // next timestep's x row
        v16h Ax  = load_afrag_f32(xrow);
        v16h Ah0 = load_afrag_h(h1h + m * H_ + laneBase);
        v16h Ah1 = load_afrag_h(h1h + m * H_ + 32 + laneBase);

        v8f acc;
#pragma unroll
        for (int r = 0; r < 8; ++r) acc[r] = bias0;
        acc = wmma_f16(Ax,  Bx,   acc);
        acc = wmma_f16(Ah0, Bh0a, acc);
        acc = wmma_f16(Ah1, Bh0b, acc);

#pragma unroll
        for (int r = 0; r < 8; ++r) gates[(laneBase + r) * G_ + n] = acc[r];
        __syncthreads();

        // -------- layer 0 cell/hidden update (branch-free activations) ------
#pragma unroll
        for (int j = 0; j < 2; ++j) {
            int hid  = hid0 + j;
            float gi = gates[mu * G_ + hid];
            float gf = gates[mu * G_ + 64 + hid];
            float gg = gates[mu * G_ + 128 + hid];
            float go = gates[mu * G_ + 192 + hid];
            c1[j] = fsig(gf) * c1[j] + fsig(gi) * ftanh(gg);
            float h = fsig(go) * ftanh(c1[j]);
            h1h[mu * H_ + hid] = (_Float16)h;
        }
        __syncthreads();

        // -------- layer 1: gates = bias + h1 @ Wih1^T + h2 @ Whh1^T --------
        v16h A2 = load_afrag_h(h1h + m * H_ + laneBase);
        v16h A3 = load_afrag_h(h1h + m * H_ + 32 + laneBase);
        v16h A4 = load_afrag_h(h2h + m * H_ + laneBase);
        v16h A5 = load_afrag_h(h2h + m * H_ + 32 + laneBase);

        v8f acc2;
#pragma unroll
        for (int r = 0; r < 8; ++r) acc2[r] = bias1;
        acc2 = wmma_f16(A2, Bi1a, acc2);
        acc2 = wmma_f16(A3, Bi1b, acc2);
        acc2 = wmma_f16(A4, Bh1a, acc2);
        acc2 = wmma_f16(A5, Bh1b, acc2);

#pragma unroll
        for (int r = 0; r < 8; ++r) gates[(laneBase + r) * G_ + n] = acc2[r];
        __syncthreads();

        // -------- layer 1 cell/hidden update --------
#pragma unroll
        for (int j = 0; j < 2; ++j) {
            int hid  = hid0 + j;
            float gi = gates[mu * G_ + hid];
            float gf = gates[mu * G_ + 64 + hid];
            float gg = gates[mu * G_ + 128 + hid];
            float go = gates[mu * G_ + 192 + hid];
            c2[j] = fsig(gf) * c2[j] + fsig(gi) * ftanh(gg);
            float h = fsig(go) * ftanh(c2[j]);
            h2h[mu * H_ + hid] = (_Float16)h;
        }
        __syncthreads();

        xrow += D_;
    }

    // -------- FC head on the final h2: out[b] = sigmoid(h2 . w_fc + b_fc) ----
    if (tid < 16) {
        float s = b_fc[0];
#pragma unroll 8
        for (int k = 0; k < H_; ++k) s += (float)h2h[tid * H_ + k] * w_fc[k];
        out[b0 + tid] = fsig(s);
    }
}

extern "C" void kernel_launch(void* const* d_in, const int* in_sizes, int n_in,
                              void* d_out, int out_size, void* d_ws, size_t ws_size,
                              hipStream_t stream) {
    (void)in_sizes; (void)n_in; (void)out_size; (void)d_ws; (void)ws_size;

    const float* x     = (const float*)d_in[0];
    const float* w_ih0 = (const float*)d_in[1];
    const float* w_hh0 = (const float*)d_in[2];
    const float* b_ih0 = (const float*)d_in[3];
    const float* b_hh0 = (const float*)d_in[4];
    const float* w_ih1 = (const float*)d_in[5];
    const float* w_hh1 = (const float*)d_in[6];
    const float* b_ih1 = (const float*)d_in[7];
    const float* b_hh1 = (const float*)d_in[8];
    const float* w_fc  = (const float*)d_in[9];
    const float* b_fc  = (const float*)d_in[10];

    lstm2_scan_kernel<<<dim3(B_ / 16), dim3(512), 0, stream>>>(
        x, w_ih0, w_hh0, b_ih0, b_hh0,
        w_ih1, w_hh1, b_ih1, b_hh1,
        w_fc, b_fc, (float*)d_out);
}
